// LRU_20005957665324
// MI455X (gfx1250) — matrix operand order repaired and checked by reading, not compiled, not run on hardware
//
#include <hip/hip_runtime.h>

// ---------------------------------------------------------------------------
// LRU forward for MI455X (gfx1250, wave32, WMMA + async-to-LDS pipelining).
//   GEMM1: Bu_{re,im}[L,H]  = x[L,Do] @ (gamma*B_{re,im})^T      (bf16 WMMA)
//   SCAN : h_t = lambda h_{t-1} + Bu_t   (chunked 3-phase, f32)
//   GEMM2: out[L,Do] = h_re@C_re^T - h_im@C_im^T + D*x           (bf16 WMMA)
// ---------------------------------------------------------------------------

#define L_SEQ  8192
#define DOUT   1024
#define DHID   2048
#define CLEN   128          // scan chunk length (= 2^7)
#define NCHUNK 64           // L_SEQ / CLEN
#define LDSTR  40           // padded LDS row stride in bf16 elements (80B = 20 banks)

typedef __bf16 bf16;
typedef bf16  bf16x16 __attribute__((ext_vector_type(16)));
typedef float f32x8   __attribute__((ext_vector_type(8)));
typedef float f32x4   __attribute__((ext_vector_type(4)));
typedef unsigned int u32x4 __attribute__((ext_vector_type(4)));

#define WAIT_ASYNC() asm volatile("s_wait_asynccnt 0x0" ::: "memory")

__device__ __forceinline__ bf16 f2bf(float f) {
  unsigned u = __builtin_bit_cast(unsigned, f);
  unsigned r = u + 0x7FFFu + ((u >> 16) & 1u);     // round-to-nearest-even
  unsigned short h = (unsigned short)(r >> 16);
  return __builtin_bit_cast(bf16, h);
}

// ---- WMMA fragment loads from LDS (contiguous k order, same in A and B) -----------
// A (16x32 bf16, MxK): lane r=lane&15 holds row M=r.
//   elements 0..15 = K[hi*16 + 0..15]
__device__ __forceinline__ bf16x16 load_frag_A(const bf16* As, int lane) {
  int r  = lane & 15;
  int hi = (lane >> 4) & 1;
  const u32x4* p0 = (const u32x4*)(As + r * LDSTR + hi * 16);
  const u32x4* p1 = (const u32x4*)(As + r * LDSTR + hi * 16 + 8);
  union { u32x4 u[2]; bf16x16 v; } cvt;
  cvt.u[0] = *p0;
  cvt.u[1] = *p1;
  return cvt.v;
}

// B (32x16 bf16, KxN) staged as rows of the N-operand (Bs[n][k]):
//   lane c=lane&15 holds column N=c; elements 0..15 = K[hi*16 + 0..15]
__device__ __forceinline__ bf16x16 load_frag_B(const bf16* Bs, int lane) {
  int c  = lane & 15;
  int hi = (lane >> 4) & 1;
  const u32x4* p = (const u32x4*)(Bs + c * LDSTR + hi * 16);
  union { u32x4 u[2]; bf16x16 v; } cvt;
  cvt.u[0] = p[0];
  cvt.u[1] = p[1];
  return cvt.v;
}

// Async-stage a 128x32 bf16 tile (global row-major) into padded LDS.
// 256 threads: thread -> (row = tid>>1, half = tid&1), 32B per thread via two
// GLOBAL_LOAD_ASYNC_TO_LDS_B128 (INST_OFFSET applies to BOTH addresses).
// Tracked with ASYNCcnt; caller must s_wait_asynccnt + barrier before use.
__device__ __forceinline__ void stage_tile_async(bf16* lds, const bf16* gsrc,
                                                 int rowStart, int kStart,
                                                 int rowStride, int tid) {
  int row  = tid >> 1;
  int half = tid & 1;
  const bf16* g = gsrc + (size_t)(rowStart + row) * rowStride + kStart + half * 16;
  unsigned l = (unsigned)(unsigned long long)(lds + row * LDSTR + half * 16);
  asm volatile("global_load_async_to_lds_b128 %0, %1, off\n\t"
               "global_load_async_to_lds_b128 %0, %1, off offset:16"
               :: "v"(l), "v"(g) : "memory");
}

// ---------------------------------------------------------------------------
// Prep: lambda, lambda^CLEN, gamma
// ---------------------------------------------------------------------------
__global__ void lru_prep(const float* __restrict__ theta_log,
                         const float* __restrict__ nu_log,
                         const float* __restrict__ gamma_log,
                         float* __restrict__ gamma,
                         float* __restrict__ lamRe, float* __restrict__ lamIm,
                         float* __restrict__ lamPRe, float* __restrict__ lamPIm) {
  int h = blockIdx.x * blockDim.x + threadIdx.x;
  if (h >= DHID) return;
  float nu = __expf(nu_log[h]);
  float th = __expf(theta_log[h]);
  float r  = __expf(-nu);
  float lr = r * __cosf(th);
  float li = r * __sinf(th);
  gamma[h] = __expf(gamma_log[h]);
  lamRe[h] = lr;
  lamIm[h] = li;
  float pr = lr, pi = li;
  #pragma unroll
  for (int s = 0; s < 7; ++s) {        // lambda^(2^7) = lambda^CLEN
    float nr = pr * pr - pi * pi;
    float ni = 2.0f * pr * pi;
    pr = nr; pi = ni;
  }
  lamPRe[h] = pr;
  lamPIm[h] = pi;
}

// ---------------------------------------------------------------------------
// f32 -> bf16 conversion kernels
// ---------------------------------------------------------------------------
__global__ void conv_x(const float* __restrict__ src, bf16* __restrict__ dst) {
  size_t i = ((size_t)blockIdx.x * blockDim.x + threadIdx.x) * 4;
  f32x4 v = *(const f32x4*)(src + i);
  dst[i + 0] = f2bf(v.x);
  dst[i + 1] = f2bf(v.y);
  dst[i + 2] = f2bf(v.z);
  dst[i + 3] = f2bf(v.w);
}

__global__ void conv_B(const float* __restrict__ src,
                       const float* __restrict__ gamma,
                       bf16* __restrict__ dst) {
  size_t i = (size_t)blockIdx.x * blockDim.x + threadIdx.x;   // [DHID][DOUT]
  float g = gamma[i >> 10];                                   // row = i / 1024
  dst[i] = f2bf(src[i] * g);
}

__global__ void conv_C(const float* __restrict__ src, bf16* __restrict__ dst,
                       float scale) {
  size_t i = (size_t)blockIdx.x * blockDim.x + threadIdx.x;
  dst[i] = f2bf(src[i] * scale);
}

// ---------------------------------------------------------------------------
// GEMM1: Bu_z[L,H] = Xb[L,1024] @ Bz[H,1024]^T  (z = blockIdx.z: 0=re, 1=im)
// 128x128 block tile, 8 waves of 32x64, K-step 32, double-buffered async LDS.
// ---------------------------------------------------------------------------
__global__ __launch_bounds__(256) void lru_gemm1(
    const bf16* __restrict__ Xb,
    const bf16* __restrict__ Bre, const bf16* __restrict__ Bim,
    float* __restrict__ BuRe, float* __restrict__ BuIm) {
  __shared__ __align__(16) bf16 As[2][128 * LDSTR];
  __shared__ __align__(16) bf16 Bs[2][128 * LDSTR];

  const int tid  = threadIdx.x;
  const int lane = tid & 31;
  const int w    = tid >> 5;
  const int wm   = (w & 3) * 32;   // 4 waves along M
  const int wn   = (w >> 2) * 64;  // 2 waves along N
  const int nBase = blockIdx.x * 128;
  const int mBase = blockIdx.y * 128;
  const bf16* Bsrc = blockIdx.z ? Bim : Bre;
  float*      Bu   = blockIdx.z ? BuIm : BuRe;

  f32x8 acc[2][4] = {};

  // prologue: stage tile 0
  stage_tile_async(As[0], Xb,   mBase, 0, DOUT, tid);
  stage_tile_async(Bs[0], Bsrc, nBase, 0, DOUT, tid);
  WAIT_ASYNC();
  __syncthreads();

  for (int kt = 0; kt < DOUT; kt += 32) {
    const int cur = (kt >> 5) & 1;
    const int nxt = cur ^ 1;
    if (kt + 32 < DOUT) {            // prefetch next tile (overlaps WMMAs)
      stage_tile_async(As[nxt], Xb,   mBase, kt + 32, DOUT, tid);
      stage_tile_async(Bs[nxt], Bsrc, nBase, kt + 32, DOUT, tid);
    }

    bf16x16 a0 = load_frag_A(As[cur] + (wm +  0) * LDSTR, lane);
    bf16x16 a1 = load_frag_A(As[cur] + (wm + 16) * LDSTR, lane);
    bf16x16 b[4];
    #pragma unroll
    for (int j = 0; j < 4; ++j)
      b[j] = load_frag_B(Bs[cur] + (wn + 16 * j) * LDSTR, lane);

    #pragma unroll
    for (int j = 0; j < 4; ++j) {
      acc[0][j] = __builtin_amdgcn_wmma_f32_16x16x32_bf16(
          false, a0, false, b[j], (short)0, acc[0][j], false, false);
      acc[1][j] = __builtin_amdgcn_wmma_f32_16x16x32_bf16(
          false, a1, false, b[j], (short)0, acc[1][j], false, false);
    }

    WAIT_ASYNC();
    __syncthreads();
  }

  const int c  = lane & 15;
  const int hi = (lane >> 4) & 1;
  #pragma unroll
  for (int i = 0; i < 2; ++i) {
    #pragma unroll
    for (int j = 0; j < 4; ++j) {
      int m0 = mBase + wm + i * 16 + hi * 8;
      int n0 = nBase + wn + j * 16 + c;
      #pragma unroll
      for (int v = 0; v < 8; ++v)
        Bu[(size_t)(m0 + v) * DHID + n0] = acc[i][j][v];
    }
  }
}

// ---------------------------------------------------------------------------
// Scan phase 1: per-chunk last value (scan from zero, no intermediates)
// ---------------------------------------------------------------------------
__global__ void lru_scan1(const float* __restrict__ BuRe,
                          const float* __restrict__ BuIm,
                          const float* __restrict__ lamRe,
                          const float* __restrict__ lamIm,
                          float* __restrict__ lastRe,
                          float* __restrict__ lastIm) {
  int h = blockIdx.y * blockDim.x + threadIdx.x;
  int c = blockIdx.x;
  float lr = lamRe[h], li = lamIm[h];
  float hr = 0.0f, hi = 0.0f;
  const float* pr = BuRe + (size_t)c * CLEN * DHID + h;
  const float* pi = BuIm + (size_t)c * CLEN * DHID + h;
  for (int t = 0; t < CLEN; ++t) {
    float br = pr[(size_t)t * DHID];
    float bi = pi[(size_t)t * DHID];
    float nr = lr * hr - li * hi + br;
    float ni = lr * hi + li * hr + bi;
    hr = nr; hi = ni;
  }
  lastRe[c * DHID + h] = hr;
  lastIm[c * DHID + h] = hi;
}

// ---------------------------------------------------------------------------
// Scan phase 2: sequential carry combine across chunks (per channel)
//   carryIn[0] = 0; carryIn[c] = lam^CLEN * carryIn[c-1] + last[c-1]
// ---------------------------------------------------------------------------
__global__ void lru_scan2(const float* __restrict__ lastRe,
                          const float* __restrict__ lastIm,
                          const float* __restrict__ lamPRe,
                          const float* __restrict__ lamPIm,
                          float* __restrict__ carRe,
                          float* __restrict__ carIm) {
  int h = blockIdx.x * blockDim.x + threadIdx.x;
  float Lr = lamPRe[h], Li = lamPIm[h];
  float cr = 0.0f, ci = 0.0f;
  for (int c = 0; c < NCHUNK; ++c) {
    carRe[c * DHID + h] = cr;
    carIm[c * DHID + h] = ci;
    float xr = lastRe[c * DHID + h];
    float xi = lastIm[c * DHID + h];
    float nr = Lr * cr - Li * ci + xr;
    float ni = Lr * ci + Li * cr + xi;
    cr = nr; ci = ni;
  }
}

// ---------------------------------------------------------------------------
// Scan phase 3: re-scan each chunk with carry, emit bf16 h and final state
// ---------------------------------------------------------------------------
__global__ void lru_scan3(const float* __restrict__ BuRe,
                          const float* __restrict__ BuIm,
                          const float* __restrict__ lamRe,
                          const float* __restrict__ lamIm,
                          const float* __restrict__ carRe,
                          const float* __restrict__ carIm,
                          bf16* __restrict__ HRe,
                          bf16* __restrict__ HIm,
                          float* __restrict__ tail) {
  int h = blockIdx.y * blockDim.x + threadIdx.x;
  int c = blockIdx.x;
  float lr = lamRe[h], li = lamIm[h];
  float hr = carRe[c * DHID + h];
  float hi = carIm[c * DHID + h];
  const size_t base = (size_t)c * CLEN * DHID + h;
  for (int t = 0; t < CLEN; ++t) {
    float br = BuRe[base + (size_t)t * DHID];
    float bi = BuIm[base + (size_t)t * DHID];
    float nr = lr * hr - li * hi + br;
    float ni = lr * hi + li * hr + bi;
    hr = nr; hi = ni;
    HRe[base + (size_t)t * DHID] = f2bf(hr);
    HIm[base + (size_t)t * DHID] = f2bf(hi);
  }
  if (c == NCHUNK - 1) {             // final hidden state, complex interleaved
    tail[2 * h + 0] = hr;
    tail[2 * h + 1] = hi;
  }
}

// ---------------------------------------------------------------------------
// GEMM2: out[L,1024] = Hre@Cre^T + Him@(-Cim)^T + D*x  (Cim pre-negated)
// Double-buffered async LDS staging of 4 tiles per K-step.
// ---------------------------------------------------------------------------
__global__ __launch_bounds__(256) void lru_gemm2(
    const bf16* __restrict__ HRe, const bf16* __restrict__ HIm,
    const bf16* __restrict__ Cre, const bf16* __restrict__ CimN,
    const float* __restrict__ X, const float* __restrict__ Dv,
    float* __restrict__ Out) {
  __shared__ __align__(16) bf16 Ar[2][128 * LDSTR];
  __shared__ __align__(16) bf16 Ai[2][128 * LDSTR];
  __shared__ __align__(16) bf16 Br[2][128 * LDSTR];
  __shared__ __align__(16) bf16 Bi[2][128 * LDSTR];

  const int tid  = threadIdx.x;
  const int lane = tid & 31;
  const int w    = tid >> 5;
  const int wm   = (w & 3) * 32;
  const int wn   = (w >> 2) * 64;
  const int nBase = blockIdx.x * 128;   // over DOUT
  const int mBase = blockIdx.y * 128;   // over L

  f32x8 acc[2][4] = {};

  stage_tile_async(Ar[0], HRe,  mBase, 0, DHID, tid);
  stage_tile_async(Ai[0], HIm,  mBase, 0, DHID, tid);
  stage_tile_async(Br[0], Cre,  nBase, 0, DHID, tid);
  stage_tile_async(Bi[0], CimN, nBase, 0, DHID, tid);
  WAIT_ASYNC();
  __syncthreads();

  for (int kt = 0; kt < DHID; kt += 32) {
    const int cur = (kt >> 5) & 1;
    const int nxt = cur ^ 1;
    if (kt + 32 < DHID) {            // prefetch next tiles (overlaps WMMAs)
      stage_tile_async(Ar[nxt], HRe,  mBase, kt + 32, DHID, tid);
      stage_tile_async(Ai[nxt], HIm,  mBase, kt + 32, DHID, tid);
      stage_tile_async(Br[nxt], Cre,  nBase, kt + 32, DHID, tid);
      stage_tile_async(Bi[nxt], CimN, nBase, kt + 32, DHID, tid);
    }

    bf16x16 ar0 = load_frag_A(Ar[cur] + (wm +  0) * LDSTR, lane);
    bf16x16 ar1 = load_frag_A(Ar[cur] + (wm + 16) * LDSTR, lane);
    bf16x16 ai0 = load_frag_A(Ai[cur] + (wm +  0) * LDSTR, lane);
    bf16x16 ai1 = load_frag_A(Ai[cur] + (wm + 16) * LDSTR, lane);
    #pragma unroll
    for (int j = 0; j < 4; ++j) {
      bf16x16 br = load_frag_B(Br[cur] + (wn + 16 * j) * LDSTR, lane);
      bf16x16 bi = load_frag_B(Bi[cur] + (wn + 16 * j) * LDSTR, lane);
      acc[0][j] = __builtin_amdgcn_wmma_f32_16x16x32_bf16(
          false, ar0, false, br, (short)0, acc[0][j], false, false);
      acc[1][j] = __builtin_amdgcn_wmma_f32_16x16x32_bf16(
          false, ar1, false, br, (short)0, acc[1][j], false, false);
      acc[0][j] = __builtin_amdgcn_wmma_f32_16x16x32_bf16(
          false, ai0, false, bi, (short)0, acc[0][j], false, false);
      acc[1][j] = __builtin_amdgcn_wmma_f32_16x16x32_bf16(
          false, ai1, false, bi, (short)0, acc[1][j], false, false);
    }

    WAIT_ASYNC();
    __syncthreads();
  }

  const int c  = lane & 15;
  const int hi = (lane >> 4) & 1;
  #pragma unroll
  for (int i = 0; i < 2; ++i) {
    #pragma unroll
    for (int j = 0; j < 4; ++j) {
      int m0 = mBase + wm + i * 16 + hi * 8;
      int n0 = nBase + wn + j * 16 + c;
      float d = Dv[n0];
      #pragma unroll
      for (int v = 0; v < 8; ++v) {
        size_t idx = (size_t)(m0 + v) * DOUT + n0;
        Out[idx] = acc[i][j][v] + d * X[idx];
      }
    }
  }
}

// ---------------------------------------------------------------------------
// Host-side orchestration
// ---------------------------------------------------------------------------
extern "C" void kernel_launch(void* const* d_in, const int* in_sizes, int n_in,
                              void* d_out, int out_size, void* d_ws, size_t ws_size,
                              hipStream_t stream) {
  const float* x         = (const float*)d_in[0];
  const float* theta_log = (const float*)d_in[1];
  const float* nu_log    = (const float*)d_in[2];
  const float* gamma_log = (const float*)d_in[3];
  const float* B_re      = (const float*)d_in[4];
  const float* B_im      = (const float*)d_in[5];
  const float* C_re      = (const float*)d_in[6];
  const float* C_im      = (const float*)d_in[7];
  const float* Dv        = (const float*)d_in[8];
  float* out  = (float*)d_out;
  float* tail = out + (size_t)L_SEQ * DOUT;   // final hidden state (re,im pairs)

  // workspace carve-out
  char* ws = (char*)d_ws;
  auto alloc = [&](size_t bytes) -> char* {
    char* p = ws;
    ws += (bytes + 255) & ~(size_t)255;
    return p;
  };
  float* gamma  = (float*)alloc((size_t)DHID * 4);
  float* lamRe  = (float*)alloc((size_t)DHID * 4);
  float* lamIm  = (float*)alloc((size_t)DHID * 4);
  float* lamPRe = (float*)alloc((size_t)DHID * 4);
  float* lamPIm = (float*)alloc((size_t)DHID * 4);
  bf16*  Xb     = (bf16*) alloc((size_t)L_SEQ * DOUT * 2);
  bf16*  Bre_b  = (bf16*) alloc((size_t)DHID * DOUT * 2);
  bf16*  Bim_b  = (bf16*) alloc((size_t)DHID * DOUT * 2);
  bf16*  Cre_b  = (bf16*) alloc((size_t)DOUT * DHID * 2);
  bf16*  CimN_b = (bf16*) alloc((size_t)DOUT * DHID * 2);
  float* BuRe   = (float*)alloc((size_t)L_SEQ * DHID * 4);
  float* BuIm   = (float*)alloc((size_t)L_SEQ * DHID * 4);
  bf16*  HRe    = (bf16*) alloc((size_t)L_SEQ * DHID * 2);
  bf16*  HIm    = (bf16*) alloc((size_t)L_SEQ * DHID * 2);
  float* lastRe = (float*)alloc((size_t)NCHUNK * DHID * 4);
  float* lastIm = (float*)alloc((size_t)NCHUNK * DHID * 4);
  float* carRe  = (float*)alloc((size_t)NCHUNK * DHID * 4);
  float* carIm  = (float*)alloc((size_t)NCHUNK * DHID * 4);

  // 0) lambda / gamma precompute
  lru_prep<<<dim3(DHID / 256), dim3(256), 0, stream>>>(
      theta_log, nu_log, gamma_log, gamma, lamRe, lamIm, lamPRe, lamPIm);

  // 1) f32 -> bf16 conversions
  conv_x<<<dim3((L_SEQ * DOUT) / (256 * 4)), dim3(256), 0, stream>>>(x, Xb);
  conv_B<<<dim3((DHID * DOUT) / 256), dim3(256), 0, stream>>>(B_re, gamma, Bre_b);
  conv_B<<<dim3((DHID * DOUT) / 256), dim3(256), 0, stream>>>(B_im, gamma, Bim_b);
  conv_C<<<dim3((DOUT * DHID) / 256), dim3(256), 0, stream>>>(C_re, Cre_b, 1.0f);
  conv_C<<<dim3((DOUT * DHID) / 256), dim3(256), 0, stream>>>(C_im, CimN_b, -1.0f);

  // 2) GEMM1 (z selects re/im B-operand)
  lru_gemm1<<<dim3(DHID / 128, L_SEQ / 128, 2), dim3(256), 0, stream>>>(
      Xb, Bre_b, Bim_b, BuRe, BuIm);

  // 3) chunked scan
  lru_scan1<<<dim3(NCHUNK, DHID / 256), dim3(256), 0, stream>>>(
      BuRe, BuIm, lamRe, lamIm, lastRe, lastIm);
  lru_scan2<<<dim3(DHID / 256), dim3(256), 0, stream>>>(
      lastRe, lastIm, lamPRe, lamPIm, carRe, carIm);
  lru_scan3<<<dim3(NCHUNK, DHID / 256), dim3(256), 0, stream>>>(
      BuRe, BuIm, lamRe, lamIm, carRe, carIm, HRe, HIm, tail);

  // 4) GEMM2 + epilogue
  lru_gemm2<<<dim3(DOUT / 128, L_SEQ / 128), dim3(256), 0, stream>>>(
      HRe, HIm, Cre_b, CimN_b, x, Dv, out);
}
